// HybridTransformerV68b_8366596292770
// MI455X (gfx1250) — matrix operand order repaired and checked by reading, not compiled, not run on hardware
//
#include <hip/hip_runtime.h>

typedef __attribute__((ext_vector_type(16))) _Float16 v16h;
typedef __attribute__((ext_vector_type(8)))  float    v8f;

#define N_BUCKETS 512
#define SLOTS_PB  32
#define DIM       512
#define T_LEN     2048
#define B_SZ      4
#define WAVES_PER_BLOCK 8

__device__ __forceinline__ float wred_add(float v) {
  #pragma unroll
  for (int m = 16; m >= 1; m >>= 1) v += __shfl_xor(v, m, 32);
  return v;
}
__device__ __forceinline__ float wred_max(float v) {
  #pragma unroll
  for (int m = 16; m >= 1; m >>= 1) v = fmaxf(v, __shfl_xor(v, m, 32));
  return v;
}

// Load 8+8 consecutive f32 (two 32B-aligned runs) and convert to v16h in the
// per-lane order: [K=h*16+0..7, K=h*16+8..15].
__device__ __forceinline__ v16h ldcvt16(const float* __restrict__ p0,
                                        const float* __restrict__ p1) {
  const float4 a = ((const float4*)p0)[0];
  const float4 b = ((const float4*)p0)[1];
  const float4 c = ((const float4*)p1)[0];
  const float4 d = ((const float4*)p1)[1];
  v16h r;
  r[0]  = (_Float16)a.x; r[1]  = (_Float16)a.y; r[2]  = (_Float16)a.z; r[3]  = (_Float16)a.w;
  r[4]  = (_Float16)b.x; r[5]  = (_Float16)b.y; r[6]  = (_Float16)b.z; r[7]  = (_Float16)b.w;
  r[8]  = (_Float16)c.x; r[9]  = (_Float16)c.y; r[10] = (_Float16)c.z; r[11] = (_Float16)c.w;
  r[12] = (_Float16)d.x; r[13] = (_Float16)d.y; r[14] = (_Float16)d.z; r[15] = (_Float16)d.w;
  return r;
}

__global__ __launch_bounds__(256) void cam_read_wmma(
    const float* __restrict__ query,
    const float* __restrict__ slot_vals,
    const float* __restrict__ slot_keys,
    const float* __restrict__ centroids,
    const int*   __restrict__ tids,
    const int*   __restrict__ slot_tids,
    float* __restrict__ out)
{
  __shared__ alignas(32) _Float16 q16[WAVES_PER_BLOCK][DIM];
  __shared__ float sc[WAVES_PER_BLOCK][SLOTS_PB];

  const int lane = threadIdx.x & 31;
  const int wv   = threadIdx.x >> 5;
  const int tok  = blockIdx.x * WAVES_PER_BLOCK + wv;   // 0..8191, exact grid
  const int b    = tok >> 11;                            // tok / 2048
  const int tid  = tids[tok];
  const int bucket = tid & (N_BUCKETS - 1);              // tid % 512 (tid >= 0)
  const int h = lane >> 4;                               // half-wave id
  const size_t slotBase = (size_t)b * (N_BUCKETS * SLOTS_PB)
                        + (size_t)bucket * SLOTS_PB;

  // ---- unified query: normalize(q), blend with anchor, normalize again ----
  // lane owns d = j*128 + lane*4 + c (fully coalesced float4 streams)
  const float* qrow = query + (size_t)tok * DIM;
  const float* crow = centroids + (size_t)bucket * DIM;
  float u[16];
  float ss = 0.f;
  #pragma unroll
  for (int j = 0; j < 4; ++j) {
    float4 v = ((const float4*)qrow)[j * 32 + lane];
    u[4*j+0] = v.x; u[4*j+1] = v.y; u[4*j+2] = v.z; u[4*j+3] = v.w;
    ss += v.x*v.x + v.y*v.y + v.z*v.z + v.w*v.w;
  }
  ss = wred_add(ss);
  const float qinv = 1.0f / fmaxf(sqrtf(ss), 1e-12f);
  float us = 0.f;
  #pragma unroll
  for (int j = 0; j < 4; ++j) {
    float4 a = ((const float4*)crow)[j * 32 + lane];
    u[4*j+0] = 0.5f * (u[4*j+0] * qinv) + 0.5f * a.x;
    u[4*j+1] = 0.5f * (u[4*j+1] * qinv) + 0.5f * a.y;
    u[4*j+2] = 0.5f * (u[4*j+2] * qinv) + 0.5f * a.z;
    u[4*j+3] = 0.5f * (u[4*j+3] * qinv) + 0.5f * a.w;
    us += u[4*j+0]*u[4*j+0] + u[4*j+1]*u[4*j+1]
        + u[4*j+2]*u[4*j+2] + u[4*j+3]*u[4*j+3];
  }
  us = wred_add(us);
  const float uinv = 1.0f / fmaxf(sqrtf(us), 1e-12f);
  #pragma unroll
  for (int j = 0; j < 4; ++j) {
    _Float16* qp = &q16[wv][j * 128 + lane * 4];
    qp[0] = (_Float16)(u[4*j+0] * uinv);
    qp[1] = (_Float16)(u[4*j+1] * uinv);
    qp[2] = (_Float16)(u[4*j+2] * uinv);
    qp[3] = (_Float16)(u[4*j+3] * uinv);
  }
  asm volatile("s_wait_dscnt 0" ::: "memory");

  // ---- scores: two 16x16x32 f16 WMMA chains over K=512 (16 steps) ----
  // A tile0: slots 0-15 (row = lane%16), A tile1: slots 16-31. B = q broadcast.
  const float* krow0 = slot_keys + (slotBase + (lane & 15)) * (size_t)DIM;
  const float* krow1 = krow0 + 16 * DIM;
  v8f c0 = {}; v8f c1 = {};
  for (int step = 0; step < 16; ++step) {
    const int d0 = step * 32 + h * 16;
    const int d1 = d0 + 8;
    v16h a0 = ldcvt16(krow0 + d0, krow0 + d1);
    v16h a1 = ldcvt16(krow1 + d0, krow1 + d1);
    v16h bq = *(const v16h*)&q16[wv][step * 32 + h * 16];
    c0 = __builtin_amdgcn_wmma_f32_16x16x32_f16(false, a0, false, bq,
                                                (short)0, c0, false, false);
    c1 = __builtin_amdgcn_wmma_f32_16x16x32_f16(false, a1, false, bq,
                                                (short)0, c1, false, false);
  }

  // ---- exchange scores so lane l owns slot l (C layout: lanes 0/16 hold n=0) ----
  if ((lane & 15) == 0) {
    #pragma unroll
    for (int r = 0; r < 8; ++r) {
      sc[wv][h * 8 + r]       = c0[r];   // tile0: slot = 8h + r
      sc[wv][16 + h * 8 + r]  = c1[r];   // tile1: slot = 16 + 8h + r
    }
  }
  asm volatile("s_wait_dscnt 0" ::: "memory");
  const float s = sc[wv][lane];

  // ---- hard/soft weight blend ----
  const int gt = slot_tids[slotBase + lane];
  const float mm = (gt == tid) ? 1.0f : 0.0f;
  const float mcnt = wred_add(mm);
  const float smax = wred_max(s);
  const float e = __expf((s - smax) * 10.0f);   // 1/TAU
  const float esum = wred_add(e);
  const bool hasm = mcnt > 0.0f;
  const float w = hasm ? (mm / (mcnt + 1e-9f)) : (e / esum);
  const float msim = hasm ? 10.0f : smax;

  // ---- weighted value sum (exact f32, coalesced streams) ----
  float acc[16];
  #pragma unroll
  for (int i = 0; i < 16; ++i) acc[i] = 0.0f;
  #pragma unroll 4
  for (int sl = 0; sl < SLOTS_PB; ++sl) {
    const float wb = __shfl(w, sl, 32);
    const float4* vr = (const float4*)(slot_vals + (slotBase + sl) * (size_t)DIM);
    #pragma unroll
    for (int j = 0; j < 4; ++j) {
      float4 v = vr[j * 32 + lane];
      acc[4*j+0] = fmaf(wb, v.x, acc[4*j+0]);
      acc[4*j+1] = fmaf(wb, v.y, acc[4*j+1]);
      acc[4*j+2] = fmaf(wb, v.z, acc[4*j+2]);
      acc[4*j+3] = fmaf(wb, v.w, acc[4*j+3]);
    }
  }

  // ---- outputs: val_out [B,T,D] then max_sim [B,T] ----
  float* orow = out + (size_t)tok * DIM;
  #pragma unroll
  for (int j = 0; j < 4; ++j) {
    float4 v;
    v.x = acc[4*j+0]; v.y = acc[4*j+1]; v.z = acc[4*j+2]; v.w = acc[4*j+3];
    ((float4*)orow)[j * 32 + lane] = v;
  }
  if (lane == 0) out[(size_t)B_SZ * T_LEN * DIM + tok] = msim;
}

extern "C" void kernel_launch(void* const* d_in, const int* in_sizes, int n_in,
                              void* d_out, int out_size, void* d_ws, size_t ws_size,
                              hipStream_t stream) {
  (void)in_sizes; (void)n_in; (void)out_size; (void)d_ws; (void)ws_size;
  const float* query = (const float*)d_in[0];
  const float* svals = (const float*)d_in[1];
  const float* skeys = (const float*)d_in[2];
  const float* cents = (const float*)d_in[3];
  const int*   tids  = (const int*)d_in[4];
  const int*   stids = (const int*)d_in[5];
  float* out = (float*)d_out;

  const int tokens = B_SZ * T_LEN;                    // 8192 waves, 1 token/wave
  dim3 grid(tokens / WAVES_PER_BLOCK);                // 1024 blocks
  dim3 block(WAVES_PER_BLOCK * 32);                   // 256 threads = 8 waves
  cam_read_wmma<<<grid, block, 0, stream>>>(query, svals, skeys, cents,
                                            tids, stids, out);
}
